// mam_80745385164821
// MI455X (gfx1250) — matrix operand order repaired and checked by reading, not compiled, not run on hardware
//
#include <hip/hip_runtime.h>
#include <hip/hip_bf16.h>

typedef _Float16 half_t;
typedef _Float16 v8h  __attribute__((ext_vector_type(8)));
typedef _Float16 v16h __attribute__((ext_vector_type(16)));
typedef float    v8f  __attribute__((ext_vector_type(8)));

#define NBATCH 16
#define NCH    3
#define IMGSZ  224
#define PATCH  16
#define HPATCH 14
#define NPATCH 196
#define PDIM   768
#define DIMV   768
#define OUTD   1024
#define DI     1536
#define TWO_DI 3072
#define NSTATE 16
#define DCONV  4
#define DTRANK 48
#define NLAYER 2
#define MROWS  (NBATCH*NPATCH)   // 3136

// ---------------------------------------------------------------------------
// Stage 0: spatial-attention gate + patch extraction -> f16 A-matrix (3136x768)
// ---------------------------------------------------------------------------
__global__ void patch_gate_kernel(const float* __restrict__ img,
                                  const float* __restrict__ sa_w,
                                  half_t* __restrict__ Xp)
{
    int idx = blockIdx.x * blockDim.x + threadIdx.x;
    if (idx >= NBATCH * IMGSZ * IMGSZ) return;
    int x = idx % IMGSZ;
    int y = (idx / IMGSZ) % IMGSZ;
    int b = idx / (IMGSZ * IMGSZ);
    const float* ib = img + (size_t)b * NCH * IMGSZ * IMGSZ;
    float acc = 0.f;
    for (int ky = 0; ky < 7; ++ky) {
        int yy = y + ky - 3;
        if (yy < 0 || yy >= IMGSZ) continue;
        for (int kx = 0; kx < 7; ++kx) {
            int xx = x + kx - 3;
            if (xx < 0 || xx >= IMGSZ) continue;
            float c0 = ib[0 * IMGSZ * IMGSZ + yy * IMGSZ + xx];
            float c1 = ib[1 * IMGSZ * IMGSZ + yy * IMGSZ + xx];
            float c2 = ib[2 * IMGSZ * IMGSZ + yy * IMGSZ + xx];
            float av = (c0 + c1 + c2) * (1.f / 3.f);
            float mx = fmaxf(c0, fmaxf(c1, c2));
            acc += av * sa_w[ky * 7 + kx] + mx * sa_w[49 + ky * 7 + kx];
        }
    }
    float g = 1.f / (1.f + expf(-acc));
    int py = y / PATCH, iy = y % PATCH, px = x / PATCH, ix = x % PATCH;
    int n = py * HPATCH + px;
    size_t rowbase = ((size_t)b * NPATCH + n) * PDIM + (size_t)(iy * PATCH + ix) * NCH;
    #pragma unroll
    for (int c = 0; c < NCH; ++c) {
        float v = ib[(size_t)c * IMGSZ * IMGSZ + y * IMGSZ + x] * g;
        Xp[rowbase + c] = (half_t)v;
    }
}

// ---------------------------------------------------------------------------
// fp32 (K,N) weight -> f16 (Npad,Kpad) transposed, zero padded
// ---------------------------------------------------------------------------
__global__ void transpose_w_kernel(const float* __restrict__ src,
                                   half_t* __restrict__ dst,
                                   int K, int N, int Kp, int Np)
{
    int idx = blockIdx.x * blockDim.x + threadIdx.x;
    if (idx >= Np * Kp) return;
    int k = idx % Kp;
    int n = idx / Kp;
    half_t v = (half_t)0.f;
    if (n < N && k < K) v = (half_t)src[(size_t)k * N + n];
    dst[(size_t)n * Kp + k] = v;
}

// ---------------------------------------------------------------------------
// WMMA GEMM: out[M,N] = A[M,Kp] (f16,row-major) x Bt[Npad,Kp]^T (f16) + bias
// 8 waves/block, 2x2 wmma tiles per wave -> 64x128 block tile. M = 3136 = 49*64.
// ---------------------------------------------------------------------------
__global__ __launch_bounds__(256) void wmma_gemm_kernel(
    const half_t* __restrict__ A, const half_t* __restrict__ Bt,
    const float* __restrict__ bias, const float* __restrict__ posemb,
    float* __restrict__ out32, half_t* __restrict__ out16,
    int Kp, int N, int ldc)
{
    const int lane = threadIdx.x & 31;
    const int wave = threadIdx.x >> 5;
    const int wm = wave & 1;
    const int wn = wave >> 1;
    const int lo = lane & 15;
    const int hi = lane >> 4;
    const int bm = blockIdx.x * 64 + wm * 32;
    const int bn = blockIdx.y * 128 + wn * 32;

    v8f acc00 = {}, acc01 = {}, acc10 = {}, acc11 = {};

    const half_t* arow0 = A + (size_t)(bm + lo) * Kp;
    const half_t* arow1 = A + (size_t)(bm + 16 + lo) * Kp;
    const half_t* brow0 = Bt + (size_t)(bn + lo) * Kp;
    const half_t* brow1 = Bt + (size_t)(bn + 16 + lo) * Kp;

    for (int k0 = 0; k0 < Kp; k0 += 32) {
        __builtin_prefetch(arow0 + k0 + 128, 0, 1);
        __builtin_prefetch(brow0 + k0 + 128, 0, 1);
        // A fragment: ISA 16-bit 16x32 layout -> two contiguous 8-half segments
        v8h a0l = *(const v8h*)(arow0 + k0 + hi * 8);
        v8h a0h = *(const v8h*)(arow0 + k0 + 16 + hi * 8);
        v8h a1l = *(const v8h*)(arow1 + k0 + hi * 8);
        v8h a1h = *(const v8h*)(arow1 + k0 + 16 + hi * 8);
        // B fragment: lane = column N, same two 8-half K segments as A from transposed weight
        v8h b0l = *(const v8h*)(brow0 + k0 + hi * 8);
        v8h b0h = *(const v8h*)(brow0 + k0 + 16 + hi * 8);
        v8h b1l = *(const v8h*)(brow1 + k0 + hi * 8);
        v8h b1h = *(const v8h*)(brow1 + k0 + 16 + hi * 8);
        v16h a0 = __builtin_shufflevector(a0l, a0h, 0,1,2,3,4,5,6,7,8,9,10,11,12,13,14,15);
        v16h a1 = __builtin_shufflevector(a1l, a1h, 0,1,2,3,4,5,6,7,8,9,10,11,12,13,14,15);
        v16h b0 = __builtin_shufflevector(b0l, b0h, 0,1,2,3,4,5,6,7,8,9,10,11,12,13,14,15);
        v16h b1 = __builtin_shufflevector(b1l, b1h, 0,1,2,3,4,5,6,7,8,9,10,11,12,13,14,15);
        acc00 = __builtin_amdgcn_wmma_f32_16x16x32_f16(false, a0, false, b0, (short)0, acc00, false, false);
        acc01 = __builtin_amdgcn_wmma_f32_16x16x32_f16(false, a0, false, b1, (short)0, acc01, false, false);
        acc10 = __builtin_amdgcn_wmma_f32_16x16x32_f16(false, a1, false, b0, (short)0, acc10, false, false);
        acc11 = __builtin_amdgcn_wmma_f32_16x16x32_f16(false, a1, false, b1, (short)0, acc11, false, false);
    }

    auto store_tile = [&](const v8f& c, int rbase, int cbase) {
        int col = cbase + lo;
        if (col >= N) return;
        float badd = bias ? bias[col] : 0.f;
        #pragma unroll
        for (int v = 0; v < 8; ++v) {
            int row = rbase + hi * 8 + v;
            float val = c[v] + badd;
            if (posemb) val += posemb[(size_t)(row % NPATCH) * ldc + col];
            if (out32) out32[(size_t)row * ldc + col] = val;
            if (out16) out16[(size_t)row * ldc + col] = (half_t)val;
        }
    };
    store_tile(acc00, bm,      bn);
    store_tile(acc01, bm,      bn + 16);
    store_tile(acc10, bm + 16, bn);
    store_tile(acc11, bm + 16, bn + 16);
}

// ---------------------------------------------------------------------------
// causal depthwise conv (DC=4) along tokens + SiLU -> f16
// ---------------------------------------------------------------------------
__global__ void conv_silu_kernel(const float* __restrict__ xr,
                                 const float* __restrict__ conv_w,
                                 const float* __restrict__ conv_b,
                                 half_t* __restrict__ xih)
{
    int idx = blockIdx.x * blockDim.x + threadIdx.x;
    if (idx >= MROWS * DI) return;
    int d = idx % DI;
    int l = (idx / DI) % NPATCH;
    int b = idx / (DI * NPATCH);
    float acc = conv_b[d];
    #pragma unroll
    for (int k = 0; k < DCONV; ++k) {
        int ls = l + k - (DCONV - 1);
        if (ls >= 0)
            acc += conv_w[d * DCONV + k] * xr[((size_t)(b * NPATCH + ls)) * TWO_DI + d];
    }
    float s = acc / (1.f + expf(-acc));
    xih[(size_t)idx] = (half_t)s;
}

// ---------------------------------------------------------------------------
// split dbc(3136x80) -> dtc f16 (3136x64 zero-padded), Bm(3136x16), Cm(3136x16)
// ---------------------------------------------------------------------------
__global__ void split_dbc_kernel(const float* __restrict__ dbc,
                                 half_t* __restrict__ dtc,
                                 float* __restrict__ Bm, float* __restrict__ Cm)
{
    int idx = blockIdx.x * blockDim.x + threadIdx.x;
    if (idx >= MROWS * 80) return;
    int c = idx % 80;
    int r = idx / 80;
    float v = dbc[idx];
    if (c < DTRANK) {
        dtc[(size_t)r * 64 + c] = (half_t)v;
    } else if (c < DTRANK + NSTATE) {
        dtc[(size_t)r * 64 + c] = (half_t)0.f;   // pad K 48->64
        Bm[(size_t)r * NSTATE + (c - DTRANK)] = v;
    } else {
        Cm[(size_t)r * NSTATE + (c - DTRANK - NSTATE)] = v;
    }
}

__global__ void softplus_kernel(float* __restrict__ x, int n)
{
    int i = blockIdx.x * blockDim.x + threadIdx.x;
    if (i >= n) return;
    float v = x[i];
    x[i] = (v > 20.f) ? v : log1pf(expf(v));
}

// ---------------------------------------------------------------------------
// selective scan: one thread per (b,d), h[16] in registers, walks L=196.
// Fuses dA, state update, C-dot, D skip, SiLU(res) gate -> y f16.
// ---------------------------------------------------------------------------
__global__ void mamba_scan_kernel(const float* __restrict__ delta,
                                  const half_t* __restrict__ xih,
                                  const float* __restrict__ xr,
                                  const float* __restrict__ Bm,
                                  const float* __restrict__ Cm,
                                  const float* __restrict__ Alog,
                                  const float* __restrict__ Dp,
                                  half_t* __restrict__ yh)
{
    int idx = blockIdx.x * blockDim.x + threadIdx.x;
    if (idx >= NBATCH * DI) return;
    int d = idx % DI;
    int b = idx / DI;
    float A[NSTATE], h[NSTATE];
    #pragma unroll
    for (int n = 0; n < NSTATE; ++n) {
        A[n] = -expf(Alog[(size_t)d * NSTATE + n]);
        h[n] = 0.f;
    }
    float Dd = Dp[d];
    for (int l = 0; l < NPATCH; ++l) {
        size_t row = (size_t)b * NPATCH + l;
        float dt  = delta[row * DI + d];
        float xi  = (float)xih[row * DI + d];
        float res = xr[row * TWO_DI + DI + d];
        float y = 0.f;
        #pragma unroll
        for (int n = 0; n < NSTATE; ++n) {
            float dA = expf(dt * A[n]);
            h[n] = dA * h[n] + dt * Bm[row * NSTATE + n] * xi;
            y += h[n] * Cm[row * NSTATE + n];
        }
        y += Dd * xi;
        float sres = res / (1.f + expf(-res));
        yh[row * DI + d] = (half_t)(y * sres);
    }
}

// ---------------------------------------------------------------------------
// per-row LayerNorm + exact GELU; optional f32/f16 outputs (in-place safe)
// ---------------------------------------------------------------------------
__global__ __launch_bounds__(256) void ln_gelu_kernel(const float* __restrict__ in,
    const float* __restrict__ g, const float* __restrict__ bt,
    float* __restrict__ out32, half_t* __restrict__ out16, int ncol)
{
    __shared__ float s1[256], s2[256];
    int row = blockIdx.x;
    const float* r = in + (size_t)row * ncol;
    float sum = 0.f, sq = 0.f;
    for (int c = threadIdx.x; c < ncol; c += 256) {
        float v = r[c];
        sum += v; sq += v * v;
    }
    s1[threadIdx.x] = sum; s2[threadIdx.x] = sq;
    __syncthreads();
    for (int off = 128; off > 0; off >>= 1) {
        if ((int)threadIdx.x < off) {
            s1[threadIdx.x] += s1[threadIdx.x + off];
            s2[threadIdx.x] += s2[threadIdx.x + off];
        }
        __syncthreads();
    }
    float mean = s1[0] / ncol;
    float var  = s2[0] / ncol - mean * mean;
    float inv  = rsqrtf(var + 1e-5f);
    for (int c = threadIdx.x; c < ncol; c += 256) {
        float v = (r[c] - mean) * inv * g[c] + bt[c];
        float ge = 0.5f * v * (1.f + erff(v * 0.70710678f));
        if (out32) out32[(size_t)row * ncol + c] = ge;
        if (out16) out16[(size_t)row * ncol + c] = (half_t)ge;
    }
}

// ---------------------------------------------------------------------------
// softmax over tokens (axis=1) fused with weighted pooling -> pooled[b,f]
// ---------------------------------------------------------------------------
__global__ void softmax_pool_kernel(const float* __restrict__ y32,
                                    const float* __restrict__ z,
                                    float* __restrict__ pooled)
{
    int idx = blockIdx.x * blockDim.x + threadIdx.x;
    if (idx >= NBATCH * OUTD) return;
    int f = idx % OUTD;
    int b = idx / OUTD;
    const float* zp = z   + (size_t)b * NPATCH * OUTD + f;
    const float* yp = y32 + (size_t)b * NPATCH * OUTD + f;
    float mx = -1e30f;
    for (int l = 0; l < NPATCH; ++l) mx = fmaxf(mx, zp[(size_t)l * OUTD]);
    float den = 0.f;
    for (int l = 0; l < NPATCH; ++l) den += expf(zp[(size_t)l * OUTD] - mx);
    float inv = 1.f / den;
    float acc = 0.f;
    for (int l = 0; l < NPATCH; ++l)
        acc += yp[(size_t)l * OUTD] * expf(zp[(size_t)l * OUTD] - mx) * inv;
    pooled[idx] = acc;
}

__global__ __launch_bounds__(256) void final_ln_kernel(const float* __restrict__ in,
    const float* __restrict__ g, const float* __restrict__ bt,
    float* __restrict__ out)
{
    __shared__ float s1[256], s2[256];
    int row = blockIdx.x;
    const float* r = in + (size_t)row * OUTD;
    float sum = 0.f, sq = 0.f;
    for (int c = threadIdx.x; c < OUTD; c += 256) {
        float v = r[c];
        sum += v; sq += v * v;
    }
    s1[threadIdx.x] = sum; s2[threadIdx.x] = sq;
    __syncthreads();
    for (int off = 128; off > 0; off >>= 1) {
        if ((int)threadIdx.x < off) {
            s1[threadIdx.x] += s1[threadIdx.x + off];
            s2[threadIdx.x] += s2[threadIdx.x + off];
        }
        __syncthreads();
    }
    float mean = s1[0] / OUTD;
    float var  = s2[0] / OUTD - mean * mean;
    float inv  = rsqrtf(var + 1e-5f);
    for (int c = threadIdx.x; c < OUTD; c += 256)
        out[(size_t)row * OUTD + c] = (r[c] - mean) * inv * g[c] + bt[c];
}

// ---------------------------------------------------------------------------
extern "C" void kernel_launch(void* const* d_in, const int* in_sizes, int n_in,
                              void* d_out, int out_size, void* d_ws, size_t ws_size,
                              hipStream_t stream) {
    const float* img      = (const float*)d_in[0];
    const float* sa_w     = (const float*)d_in[1];
    const float* pe_w     = (const float*)d_in[2];
    const float* pe_b     = (const float*)d_in[3];
    const float* pos_emb  = (const float*)d_in[4];
    const float* m_in_w   = (const float*)d_in[5];
    const float* m_in_b   = (const float*)d_in[6];
    const float* m_conv_w = (const float*)d_in[7];
    const float* m_conv_b = (const float*)d_in[8];
    const float* m_xproj  = (const float*)d_in[9];
    const float* m_dt_w   = (const float*)d_in[10];
    const float* m_dt_b   = (const float*)d_in[11];
    const float* m_Alog   = (const float*)d_in[12];
    const float* m_D      = (const float*)d_in[13];
    const float* m_out_w  = (const float*)d_in[14];
    const float* m_out_b  = (const float*)d_in[15];
    const float* em_w     = (const float*)d_in[16];
    const float* em_b     = (const float*)d_in[17];
    const float* em_g     = (const float*)d_in[18];
    const float* em_bt    = (const float*)d_in[19];
    const float* att_w    = (const float*)d_in[20];
    const float* att_b    = (const float*)d_in[21];
    const float* att_g    = (const float*)d_in[22];
    const float* att_bt   = (const float*)d_in[23];
    const float* lat_g    = (const float*)d_in[24];
    const float* lat_bt   = (const float*)d_in[25];
    float* out = (float*)d_out;

    char* ws = (char*)d_ws;
    size_t off = 0;
    auto alloc = [&](size_t bytes) {
        off = (off + 255) & ~(size_t)255;
        size_t o = off;
        off += bytes;
        return o;
    };
    half_t* Xp   = (half_t*)(ws + alloc((size_t)MROWS * PDIM * 2));
    half_t* xh   = (half_t*)(ws + alloc((size_t)MROWS * DIMV * 2));
    half_t* xih  = (half_t*)(ws + alloc((size_t)MROWS * DI * 2));
    size_t  xr_o = alloc((size_t)MROWS * TWO_DI * 4);
    float*  xr   = (float*)(ws + xr_o);
    float*  dbc  = (float*)(ws + alloc((size_t)MROWS * 80 * 4));
    half_t* dtc  = (half_t*)(ws + alloc((size_t)MROWS * 64 * 2));
    float*  Bm   = (float*)(ws + alloc((size_t)MROWS * NSTATE * 4));
    float*  Cm   = (float*)(ws + alloc((size_t)MROWS * NSTATE * 4));
    float*  delt = (float*)(ws + alloc((size_t)MROWS * DI * 4));
    half_t* yh   = (half_t*)(ws + alloc((size_t)MROWS * DI * 2));
    half_t* wt   = (half_t*)(ws + alloc((size_t)TWO_DI * DIMV * 2));
    float*  y32  = (float*)(ws + alloc((size_t)MROWS * OUTD * 4));
    half_t* yh2  = (half_t*)(ws + alloc((size_t)MROWS * OUTD * 2));
    float*  pool = (float*)(ws + alloc((size_t)NBATCH * OUTD * 4));
    // alias post-mamba f32 activations into xr (dead after last out-proj GEMM)
    float*  e32   = (float*)(ws + xr_o);
    float*  att32 = (float*)(ws + xr_o + (16u << 20));

    auto tpose = [&](const float* src, int K, int N, int Kp, int Np) {
        int total = Np * Kp;
        transpose_w_kernel<<<(total + 255) / 256, 256, 0, stream>>>(src, wt, K, N, Kp, Np);
    };
    auto gemm = [&](const half_t* A, const float* bias, const float* pos,
                    float* o32, half_t* o16, int Kp, int N) {
        int Npad = (N + 127) / 128 * 128;
        dim3 grid(MROWS / 64, Npad / 128);
        wmma_gemm_kernel<<<grid, 256, 0, stream>>>(A, wt, bias, pos, o32, o16, Kp, N, N);
    };

    // Stage 0: gate + patchify
    patch_gate_kernel<<<(NBATCH * IMGSZ * IMGSZ + 255) / 256, 256, 0, stream>>>(img, sa_w, Xp);

    // Patch embed + pos_emb (fused) -> x f16
    tpose(pe_w, PDIM, DIMV, PDIM, DIMV);
    gemm(Xp, pe_b, pos_emb, nullptr, xh, PDIM, DIMV);

    for (int i = 0; i < NLAYER; ++i) {
        const float* in_w   = m_in_w   + (size_t)i * DIMV * TWO_DI;
        const float* in_b   = m_in_b   + (size_t)i * TWO_DI;
        const float* conv_w = m_conv_w + (size_t)i * DI * DCONV;
        const float* conv_b = m_conv_b + (size_t)i * DI;
        const float* xproj  = m_xproj  + (size_t)i * DI * 80;
        const float* dt_w   = m_dt_w   + (size_t)i * DTRANK * DI;
        const float* dt_b   = m_dt_b   + (size_t)i * DI;
        const float* Alog   = m_Alog   + (size_t)i * DI * NSTATE;
        const float* Dp     = m_D      + (size_t)i * DI;
        const float* out_w  = m_out_w  + (size_t)i * DI * DIMV;
        const float* out_b  = m_out_b  + (size_t)i * DIMV;

        // in-proj: xr = x @ in_w + in_b   (3136 x 3072, f32)
        tpose(in_w, DIMV, TWO_DI, DIMV, TWO_DI);
        gemm(xh, in_b, nullptr, xr, nullptr, DIMV, TWO_DI);

        // causal depthwise conv + SiLU -> xi f16
        conv_silu_kernel<<<(MROWS * DI + 255) / 256, 256, 0, stream>>>(xr, conv_w, conv_b, xih);

        // x-proj: dbc = xi @ xproj_w   (3136 x 80)
        tpose(xproj, DI, 80, DI, 128);
        gemm(xih, nullptr, nullptr, dbc, nullptr, DI, 80);

        split_dbc_kernel<<<(MROWS * 80 + 255) / 256, 256, 0, stream>>>(dbc, dtc, Bm, Cm);

        // delta = softplus(dtc @ dt_w + dt_b)   (K padded 48->64)
        tpose(dt_w, DTRANK, DI, 64, DI);
        gemm(dtc, dt_b, nullptr, delt, nullptr, 64, DI);
        softplus_kernel<<<(MROWS * DI + 255) / 256, 256, 0, stream>>>(delt, MROWS * DI);

        // selective scan + D skip + SiLU(res) gate -> y f16
        mamba_scan_kernel<<<(NBATCH * DI + 255) / 256, 256, 0, stream>>>(
            delt, xih, xr, Bm, Cm, Alog, Dp, yh);

        // out-proj -> new x f16
        tpose(out_w, DI, DIMV, DI, DIMV);
        gemm(yh, out_b, nullptr, nullptr, xh, DI, DIMV);
    }

    // em head: e = x @ em_w + em_b ; y = gelu(ln(e))
    tpose(em_w, DIMV, OUTD, DIMV, OUTD);
    gemm(xh, em_b, nullptr, e32, nullptr, DIMV, OUTD);
    ln_gelu_kernel<<<MROWS, 256, 0, stream>>>(e32, em_g, em_bt, y32, yh2, OUTD);

    // att head: z_pre = gelu(ln(y @ att_w + att_b))
    tpose(att_w, OUTD, OUTD, OUTD, OUTD);
    gemm(yh2, att_b, nullptr, att32, nullptr, OUTD, OUTD);
    ln_gelu_kernel<<<MROWS, 256, 0, stream>>>(att32, att_g, att_bt, att32, nullptr, OUTD);

    // softmax over tokens + weighted pool, then final LN
    softmax_pool_kernel<<<(NBATCH * OUTD + 255) / 256, 256, 0, stream>>>(y32, att32, pool);
    final_ln_kernel<<<NBATCH, 256, 0, stream>>>(pool, lat_g, lat_bt, out);
}
